// VAEAttentionBlock_4492535792575
// MI455X (gfx1250) — matrix operand order repaired and checked by reading, not compiled, not run on hardware
//
#include <hip/hip_runtime.h>
#include <cstdint>
#include <cstddef>

#define NSP  4096   // H*W tokens
#define CCH  512    // channels / head dim
#define NB   2      // batch
#define NGRP 32     // groupnorm groups

typedef __attribute__((ext_vector_type(16))) __bf16 v16bf;
typedef __attribute__((ext_vector_type(8)))  float  v8f;
typedef unsigned int u32;
typedef u32 u32x4 __attribute__((ext_vector_type(4)));

union Frag { u32x4 u[2]; v16bf v; };

__device__ __forceinline__ uint16_t f2bf(float f) {
  u32 u = __float_as_uint(f);
  u32 r = u + 0x7FFFu + ((u >> 16) & 1u);   // round-to-nearest-even
  return (uint16_t)(r >> 16);
}

__device__ __forceinline__ v8f wmma_bf16(const Frag& a, const Frag& b, v8f c) {
  return __builtin_amdgcn_wmma_f32_16x16x32_bf16(false, a.v, false, b.v,
                                                 (short)0, c, false, false);
}

// ---------------- GroupNorm stats: one block per (b, group) ----------------
__global__ __launch_bounds__(256)
void gn_stats_kernel(const float* __restrict__ x, float* __restrict__ mean,
                     float* __restrict__ rstd) {
  const int bg = blockIdx.x;                 // 0..63
  const int b = bg >> 5, g = bg & 31;
  const float* base = x + (size_t)b * CCH * NSP + (size_t)g * 16 * NSP;
  float s = 0.f, ss = 0.f;
  for (int i = threadIdx.x; i < 16 * NSP; i += 256) {
    float v = base[i];
    s += v; ss += v * v;
  }
  __shared__ float rs[256], rss[256];
  rs[threadIdx.x] = s; rss[threadIdx.x] = ss;
  __syncthreads();
  for (int st = 128; st > 0; st >>= 1) {
    if (threadIdx.x < st) {
      rs[threadIdx.x]  += rs[threadIdx.x + st];
      rss[threadIdx.x] += rss[threadIdx.x + st];
    }
    __syncthreads();
  }
  if (threadIdx.x == 0) {
    const float inv = 1.f / (16.f * NSP);
    float mu = rs[0] * inv;
    float var = rss[0] * inv - mu * mu;
    mean[bg] = mu;
    rstd[bg] = rsqrtf(var + 1e-5f);
  }
}

// ---------- GroupNorm apply + transpose to bf16 tokens [B, N, C] ----------
__global__ __launch_bounds__(256)
void gn_apply_kernel(const float* __restrict__ x, const float* __restrict__ mean,
                     const float* __restrict__ rstd, const float* __restrict__ gw,
                     const float* __restrict__ gb, uint16_t* __restrict__ tkn) {
  size_t idx = (size_t)blockIdx.x * 256 + threadIdx.x;  // over [b, n, c], c fastest
  int c = (int)(idx & (CCH - 1));
  size_t bn = idx >> 9;
  int n = (int)(bn & (NSP - 1));
  int b = (int)(bn >> 12);
  int g = c >> 4;
  float mu = mean[b * NGRP + g];
  float rs = rstd[b * NGRP + g];
  float v = x[((size_t)(b * CCH + c) << 12) + n];
  tkn[idx] = f2bf((v - mu) * rs * gw[c] + gb[c]);
}

// ---------------------------- f32 -> bf16 copy ----------------------------
__global__ __launch_bounds__(256)
void cvt_kernel(const float* __restrict__ src, uint16_t* __restrict__ dst, int n) {
  int i = blockIdx.x * 256 + threadIdx.x;
  if (i < n) dst[i] = f2bf(src[i]);
}

// -------- Tiled WMMA GEMM: out = A[M,K] @ W[N,K]^T + bias, 3 modes --------
// mode 0: bf16 out[m*Nn + n]
// mode 1: bf16 out[(b*Nn + n)*NSP + nn]                (V stored transposed)
// mode 2: f32  out[(b*Nn + n)*NSP + nn] + resid[same]  (final proj+residual)
__global__ __launch_bounds__(256)
void gemm_bf16_kernel(const uint16_t* __restrict__ A, const uint16_t* __restrict__ W,
                      const float* __restrict__ bias,
                      uint16_t* __restrict__ outb, float* __restrict__ outf,
                      const float* __restrict__ resid,
                      int M, int Nn, int K, int mode) {
  const int lane = threadIdx.x & 31;
  const int w    = threadIdx.x >> 5;
  const int li   = lane & 15;
  const int hi   = lane >> 4;
  const int ntiles = Nn >> 7;
  const int bm = blockIdx.x / ntiles;
  const int bn = blockIdx.x % ntiles;
  const int wm = w >> 1;                 // 0..3
  const int wn = w & 1;                  // 0..1
  const int row0 = bm * 128 + wm * 32;
  const int col0 = bn * 128 + wn * 64;

  v8f acc[2][4];
  #pragma unroll
  for (int rt = 0; rt < 2; ++rt)
    #pragma unroll
    for (int ct = 0; ct < 4; ++ct)
      #pragma unroll
      for (int r = 0; r < 8; ++r) acc[rt][ct][r] = 0.f;

  for (int k0 = 0; k0 < K; k0 += 32) {
    Frag a[2];
    #pragma unroll
    for (int rt = 0; rt < 2; ++rt) {
      const uint16_t* p = A + (size_t)(row0 + rt * 16 + li) * K + k0 + hi * 8;
      a[rt].u[0] = *(const u32x4*)p;
      a[rt].u[1] = *(const u32x4*)(p + 16);
    }
    Frag bb[4];
    #pragma unroll
    for (int ct = 0; ct < 4; ++ct) {
      const uint16_t* p = W + (size_t)(col0 + ct * 16 + li) * K + k0 + hi * 8;
      bb[ct].u[0] = *(const u32x4*)p;
      bb[ct].u[1] = *(const u32x4*)(p + 16);
    }
    #pragma unroll
    for (int rt = 0; rt < 2; ++rt)
      #pragma unroll
      for (int ct = 0; ct < 4; ++ct)
        acc[rt][ct] = wmma_bf16(a[rt], bb[ct], acc[rt][ct]);
  }

  float bvv[4];
  #pragma unroll
  for (int ct = 0; ct < 4; ++ct) bvv[ct] = bias[col0 + ct * 16 + li];

  #pragma unroll
  for (int rt = 0; rt < 2; ++rt) {
    #pragma unroll
    for (int ct = 0; ct < 4; ++ct) {
      const int n = col0 + ct * 16 + li;
      #pragma unroll
      for (int r = 0; r < 8; ++r) {
        const int m = row0 + rt * 16 + hi * 8 + r;
        const float v = acc[rt][ct][r] + bvv[ct];
        if (mode == 0) {
          outb[(size_t)m * Nn + n] = f2bf(v);
        } else {
          const int b2 = m >> 12;
          const int nn = m & (NSP - 1);
          const size_t idx = ((size_t)(b2 * Nn + n) << 12) + nn;
          if (mode == 1) outb[idx] = f2bf(v);
          else           outf[idx] = v + resid[idx];
        }
      }
    }
  }
}

// ------------------- Flash attention, 32-query blocks ---------------------
__global__ __launch_bounds__(256)
void attn_kernel(const uint16_t* __restrict__ Q, const uint16_t* __restrict__ Kk,
                 const uint16_t* __restrict__ Vt, uint16_t* __restrict__ O) {
  __shared__ float    Sa[32 * 32];
  __shared__ float    Sb[32 * 32];
  __shared__ uint16_t Pm[32 * 32];
  __shared__ float    sfrow[32];
  __shared__ float    lrow[32];

  const int lane = threadIdx.x & 31;
  const int w    = threadIdx.x >> 5;
  const int li   = lane & 15;
  const int hi   = lane >> 4;
  const int b    = blockIdx.x >> 7;
  const int qb   = blockIdx.x & 127;
  const int q0   = qb * 32;
  const uint16_t* Qp = Q  + (size_t)b * NSP * CCH;
  const uint16_t* Kp = Kk + (size_t)b * NSP * CCH;
  const uint16_t* Vp = Vt + (size_t)b * CCH * NSP;

  const int srt = w & 1, sct = (w >> 1) & 1, shalf = w >> 2;  // S roles
  const int ort = w & 1, ocg = w >> 1;                        // O roles

  // preload Q fragments for this wave's K-half (reused over all key blocks)
  Frag qf[8];
  #pragma unroll
  for (int s = 0; s < 8; ++s) {
    const int k0 = shalf * 256 + s * 32;
    const uint16_t* p = Qp + (size_t)(q0 + srt * 16 + li) * CCH + k0 + hi * 8;
    qf[s].u[0] = *(const u32x4*)p;
    qf[s].u[1] = *(const u32x4*)(p + 16);
  }

  v8f o[8];
  #pragma unroll
  for (int t = 0; t < 8; ++t)
    #pragma unroll
    for (int r = 0; r < 8; ++r) o[t][r] = 0.f;

  float m_i = -__builtin_inff(), l_i = 0.f;
  const float scale = 0.044194173824159216f;  // 512^-0.5

  for (int kb = 0; kb < NSP / 32; ++kb) {
    const int j0 = kb * 32;

    // Prefetch next key block's K rows and V rows into the WGP cache while the
    // barrier-bounded softmax/PV section below runs (global_prefetch_b8; no
    // counters, safe across barriers). Each lane touches a distinct row's line.
    if (kb + 1 < NSP / 32) {
      const int j1 = j0 + 32;
      // next K tile: 32 rows, row stride CCH*2 bytes; 8 waves x 32 lanes cover it
      __builtin_prefetch(Kp + (size_t)(j1 + (w & 3) * 8 + (lane >> 2)) * CCH +
                             (lane & 3) * 128, 0, 0);
      // next V tile columns for this wave's d-range (ocg*128 .. +127)
      __builtin_prefetch(Vp + (size_t)(ocg * 128 + (w >> 2) * 64 + (lane << 1)) * NSP + j1,
                         0, 0);
    }

    // ---- partial S = Q K^T over this wave's K-half ----
    v8f s;
    #pragma unroll
    for (int r = 0; r < 8; ++r) s[r] = 0.f;
    #pragma unroll
    for (int st = 0; st < 8; ++st) {
      const int k0 = shalf * 256 + st * 32;
      Frag kf;
      const uint16_t* p = Kp + (size_t)(j0 + sct * 16 + li) * CCH + k0 + hi * 8;
      kf.u[0] = *(const u32x4*)p;
      kf.u[1] = *(const u32x4*)(p + 16);
      s = wmma_bf16(qf[st], kf, s);
    }
    float* Sdst = shalf ? Sb : Sa;
    #pragma unroll
    for (int r = 0; r < 8; ++r)
      Sdst[(srt * 16 + hi * 8 + r) * 32 + sct * 16 + li] = s[r];
    __syncthreads();

    // ---- online softmax row pass (wave 0, one lane per row) ----
    if (threadIdx.x < 32) {
      const int i = threadIdx.x;
      float v[32];
      float mx = m_i;
      #pragma unroll
      for (int j = 0; j < 32; ++j) {
        float t = (Sa[i * 32 + j] + Sb[i * 32 + j]) * scale;
        v[j] = t;
        mx = fmaxf(mx, t);
      }
      float sum = 0.f;
      #pragma unroll
      for (int j = 0; j < 32; ++j) {
        float e = __expf(v[j] - mx);
        sum += e;
        Pm[i * 32 + j] = f2bf(e);
      }
      const float sf = __expf(m_i - mx);
      l_i = l_i * sf + sum;
      m_i = mx;
      sfrow[i] = sf;
    }
    __syncthreads();

    // ---- rescale O, then O += P V ----
    Frag pf;
    {
      const uint16_t* pp = Pm + (ort * 16 + li) * 32 + hi * 8;
      pf.u[0] = *(const u32x4*)pp;
      pf.u[1] = *(const u32x4*)(pp + 16);
    }
    float sfv[8];
    #pragma unroll
    for (int r = 0; r < 8; ++r) sfv[r] = sfrow[ort * 16 + hi * 8 + r];
    #pragma unroll
    for (int t = 0; t < 8; ++t) {
      #pragma unroll
      for (int r = 0; r < 8; ++r) o[t][r] *= sfv[r];
      Frag vf;
      const uint16_t* p = Vp + (size_t)(ocg * 128 + t * 16 + li) * NSP + j0 + hi * 8;
      vf.u[0] = *(const u32x4*)p;
      vf.u[1] = *(const u32x4*)(p + 16);
      o[t] = wmma_bf16(pf, vf, o[t]);
    }
    __syncthreads();
  }

  // ---- finalize: divide by row sums, store bf16 [B, N, C] ----
  if (threadIdx.x < 32) lrow[threadIdx.x] = l_i;
  __syncthreads();
  float linv[8];
  #pragma unroll
  for (int r = 0; r < 8; ++r) linv[r] = 1.f / lrow[ort * 16 + hi * 8 + r];
  #pragma unroll
  for (int t = 0; t < 8; ++t) {
    const int col = ocg * 128 + t * 16 + li;
    #pragma unroll
    for (int r = 0; r < 8; ++r) {
      const int m = q0 + ort * 16 + hi * 8 + r;
      O[(size_t)b * NSP * CCH + (size_t)m * CCH + col] = f2bf(o[t][r] * linv[r]);
    }
  }
}

extern "C" void kernel_launch(void* const* d_in, const int* in_sizes, int n_in,
                              void* d_out, int out_size, void* d_ws, size_t ws_size,
                              hipStream_t stream) {
  (void)in_sizes; (void)n_in; (void)out_size; (void)ws_size;
  const float* x  = (const float*)d_in[0];
  const float* gw = (const float*)d_in[1];
  const float* gb = (const float*)d_in[2];
  const float* wq = (const float*)d_in[3];
  const float* bq = (const float*)d_in[4];
  const float* wk = (const float*)d_in[5];
  const float* bk = (const float*)d_in[6];
  const float* wv = (const float*)d_in[7];
  const float* bv = (const float*)d_in[8];
  const float* wo = (const float*)d_in[9];
  const float* bo = (const float*)d_in[10];
  float* out = (float*)d_out;

  uint8_t* wsb = (uint8_t*)d_ws;
  size_t off = 0;
  auto take = [&](size_t bytes) -> void* {
    void* p = wsb + off;
    off = (off + bytes + 255) & ~(size_t)255;
    return p;
  };
  const size_t tok = (size_t)NB * NSP * CCH;            // token elements
  float*    mean  = (float*)take(NB * NGRP * sizeof(float));
  float*    rstd  = (float*)take(NB * NGRP * sizeof(float));
  uint16_t* t_bf  = (uint16_t*)take(tok * 2);
  uint16_t* wq_bf = (uint16_t*)take((size_t)CCH * CCH * 2);
  uint16_t* wk_bf = (uint16_t*)take((size_t)CCH * CCH * 2);
  uint16_t* wv_bf = (uint16_t*)take((size_t)CCH * CCH * 2);
  uint16_t* wo_bf = (uint16_t*)take((size_t)CCH * CCH * 2);
  uint16_t* Qb    = (uint16_t*)take(tok * 2);
  uint16_t* Kb    = (uint16_t*)take(tok * 2);
  uint16_t* Vtb   = (uint16_t*)take(tok * 2);
  uint16_t* Ab    = (uint16_t*)take(tok * 2);

  // 1) GroupNorm
  gn_stats_kernel<<<NB * NGRP, 256, 0, stream>>>(x, mean, rstd);
  gn_apply_kernel<<<(int)(tok / 256), 256, 0, stream>>>(x, mean, rstd, gw, gb, t_bf);

  // 2) weights -> bf16
  const int wcnt = CCH * CCH;
  cvt_kernel<<<wcnt / 256, 256, 0, stream>>>(wq, wq_bf, wcnt);
  cvt_kernel<<<wcnt / 256, 256, 0, stream>>>(wk, wk_bf, wcnt);
  cvt_kernel<<<wcnt / 256, 256, 0, stream>>>(wv, wv_bf, wcnt);
  cvt_kernel<<<wcnt / 256, 256, 0, stream>>>(wo, wo_bf, wcnt);

  // 3) Q/K/V projections (M = B*N = 8192, Nn = K = 512); V stored transposed
  const int M = NB * NSP;
  const int gemm_grid = (M / 128) * (CCH / 128);  // 256 blocks
  gemm_bf16_kernel<<<gemm_grid, 256, 0, stream>>>(t_bf, wq_bf, bq, Qb, nullptr, nullptr, M, CCH, CCH, 0);
  gemm_bf16_kernel<<<gemm_grid, 256, 0, stream>>>(t_bf, wk_bf, bk, Kb, nullptr, nullptr, M, CCH, CCH, 0);
  gemm_bf16_kernel<<<gemm_grid, 256, 0, stream>>>(t_bf, wv_bf, bv, Vtb, nullptr, nullptr, M, CCH, CCH, 1);

  // 4) flash attention, 32 queries per block
  attn_kernel<<<NB * (NSP / 32), 256, 0, stream>>>(Qb, Kb, Vtb, Ab);

  // 5) output projection + bias + residual, back to [B, C, H, W] f32
  gemm_bf16_kernel<<<gemm_grid, 256, 0, stream>>>(Ab, wo_bf, bo, nullptr, out, x, M, CCH, CCH, 2);
}
